// Memory_85718957294295
// MI455X (gfx1250) — matrix operand order repaired and checked by reading, not compiled, not run on hardware
//
#include <hip/hip_runtime.h>
#include <hip/hip_bf16.h>
#include <stdint.h>

// ---------------------------------------------------------------------------
// MI455X (gfx1250): bf16 WMMA for all three GEMMs, fused streaming top-128.
// Round 3: bank-skewed A-tile (520-elem stride, conflict-free ds_loads),
// TDM tensor_load_to_lds with hardware pad for the skew, s_wait_tensorcnt.
// ---------------------------------------------------------------------------

typedef __bf16 bf16_t;
typedef __attribute__((ext_vector_type(16))) __bf16 v16bf;
typedef __attribute__((ext_vector_type(8)))  __bf16 v8bf;
typedef __attribute__((ext_vector_type(8)))  float  v8f;
typedef __attribute__((ext_vector_type(4)))  unsigned int u32x4;
typedef __attribute__((ext_vector_type(8)))  int i32x8;
typedef __attribute__((ext_vector_type(4)))  int i32x4;

#define T_TOK 4096   // B*L
#define DFEAT 1024   // D
#define PDIM  512    // P
#define NPAT  65536  // N
#define KTOP  128

// A-tile row stride in bf16 elements: 512 data + 8 skew (16 B) so that
// consecutive rows differ by 260 dwords == 4 (mod 64 banks) -> conflict-free.
#define A_STR     520
#define A_STR_U32 260

#define SHUF16(lo, hi) __builtin_shufflevector((lo), (hi), 0,1,2,3,4,5,6,7,8,9,10,11,12,13,14,15)

// ---------------------------------------------------------------------------
// elementwise f32 -> bf16 cast
// ---------------------------------------------------------------------------
__global__ void __launch_bounds__(256)
cast_f32_to_bf16(const float* __restrict__ in, bf16_t* __restrict__ out, int n) {
    int i = blockIdx.x * 256 + threadIdx.x;
    if (i < n) out[i] = (bf16_t)in[i];
}

// ---------------------------------------------------------------------------
// row L2-normalize (torch cosine eps clamp) -> bf16.  One block per row.
// ---------------------------------------------------------------------------
__global__ void __launch_bounds__(256)
rownorm_to_bf16(const float* __restrict__ in, bf16_t* __restrict__ out, int cols) {
    __shared__ float red[256];
    const int row = blockIdx.x;
    const float* r = in + (size_t)row * cols;
    float s = 0.f;
    for (int c = threadIdx.x; c < cols; c += 256) { float v = r[c]; s += v * v; }
    red[threadIdx.x] = s;
    __syncthreads();
    for (int o = 128; o > 0; o >>= 1) {
        if (threadIdx.x < o) red[threadIdx.x] += red[threadIdx.x + o];
        __syncthreads();
    }
    const float inv = 1.0f / fmaxf(sqrtf(red[0]), 1e-8f);
    for (int c = threadIdx.x; c < cols; c += 256)
        out[(size_t)row * cols + c] = (bf16_t)(r[c] * inv);
}

// ---------------------------------------------------------------------------
// encoder GEMM: enc[T,P] = x_bf16[T,D] @ enc_w_bf16[P,D]^T + enc_b  (f32 out)
// ---------------------------------------------------------------------------
__global__ void __launch_bounds__(128)
enc_gemm_kernel(const bf16_t* __restrict__ xbf, const bf16_t* __restrict__ wbf,
                const float* __restrict__ bias, float* __restrict__ enc) {
    const int tid  = threadIdx.x;
    const int wave = tid >> 5, lane = tid & 31;
    const int half = lane >> 4, l15 = lane & 15;
    const int tM = blockIdx.x * 16;
    const int tN = blockIdx.y * 64 + wave * 16;

    const bf16_t* arow = xbf + (size_t)(tM + l15) * DFEAT;
    const bf16_t* brow = wbf + (size_t)(tN + l15) * DFEAT + half * 16;

    v8f c = {};
#pragma unroll 4
    for (int k0 = 0; k0 < DFEAT; k0 += 32) {
        v8bf alo = *(const v8bf*)(arow + k0 + half * 16);
        v8bf ahi = *(const v8bf*)(arow + k0 + half * 16 + 8);
        v16bf a = SHUF16(alo, ahi);
        v16bf b = *(const v16bf*)(brow + k0);
        c = __builtin_amdgcn_wmma_f32_16x16x32_bf16(false, a, false, b,
                                                    (short)0, c, false, false);
    }
    const int n = tN + l15;
    const float bn = bias[n];
#pragma unroll
    for (int r = 0; r < 8; ++r) {
        const int m = tM + r + 8 * half;
        enc[(size_t)m * PDIM + n] = c[r] + bn;
    }
}

// ---------------------------------------------------------------------------
// fused similarity GEMM + streaming top-128.
// block = 128 threads (4 waves); 32 tokens per block; 128 patterns per step.
// Each wave computes a 2Mx2N register tile (four independent accumulators).
// A-tile staged by the Tensor Data Mover with hardware padding that realizes
// the bank-skewed 520-element row stride.
// ---------------------------------------------------------------------------
__global__ void __launch_bounds__(128)
sim_topk_kernel(const bf16_t* __restrict__ encn, const bf16_t* __restrict__ patn,
                int* __restrict__ topk) {
    __shared__ bf16_t   aTile[32 * A_STR];     // 32.5 KB (skewed rows)
    __shared__ float    topVal[32 * KTOP];     // 16 KB
    __shared__ int      topIdx[32 * KTOP];     // 16 KB
    __shared__ float    simBuf[32 * 128];      // 16 KB
    __shared__ float    thresh[32];
    __shared__ int      minPos[32];
    __shared__ unsigned passMask[32 * 4];

    const int tid  = threadIdx.x;
    const int wave = tid >> 5, lane = tid & 31;
    const int half = lane >> 4, l15 = lane & 15;
    const int tM = blockIdx.x * 32;

#if defined(__gfx1250__) && __has_builtin(__builtin_amdgcn_tensor_load_to_lds)
    // ---- TDM: async DMA of the 32 x 512 bf16 tile into skewed LDS rows ----
    if (wave == 0) {
        const unsigned long long ga =
            (unsigned long long)(uintptr_t)(encn + (size_t)tM * PDIM);
        const unsigned ldsOff = (unsigned)(uintptr_t)(&aTile[0]);
        u32x4 g0;
        g0[0] = 1u;                                   // count=1 valid descriptor
        g0[1] = ldsOff;                               // lds_addr (bytes)
        g0[2] = (unsigned)ga;                         // global_addr[31:0]
        g0[3] = (unsigned)((ga >> 32) & 0x01FFFFFFu)  // global_addr[56:32]
              | (2u << 30);                           // type = 2 ("image")
        i32x8 g1;
        // data_size=2B | pad_enable | pad_interval=256 dwords | pad_amount=4 dwords
        g1[0] = (int)((1u << 16) | (1u << 20) | (7u << 22) | (3u << 25));
        g1[1] = (int)(512u << 16);   // tensor_dim0 = 512 (bits 79:48, low half)
        g1[2] = (int)(32u << 16);    // tensor_dim1 = 32  (bits 111:80, low half)
        g1[3] = (int)(512u << 16);   // tile_dim0 = 512   (bits 127:112)
        g1[4] = 32;                  // tile_dim1 = 32    (bits 143:128)
        g1[5] = 512;                 // tensor_dim0_stride = 512 (bits 207:160)
        g1[6] = 0;
        g1[7] = 0;
        i32x4 gz = {0, 0, 0, 0};
#if __clang_major__ >= 23
        i32x8 gz8 = {0, 0, 0, 0, 0, 0, 0, 0};
        __builtin_amdgcn_tensor_load_to_lds(g0, g1, gz, gz, gz8, 0);
#else
        __builtin_amdgcn_tensor_load_to_lds(g0, g1, gz, gz, 0);
#endif
    }
#else
    // ---- fallback: manual copy with the same bank-skewed stride ----
    {
        const unsigned* src = (const unsigned*)(encn + (size_t)tM * PDIM);
        unsigned* dst = (unsigned*)aTile;
        for (int i = tid; i < 32 * 256; i += 128) {
            const int row = i >> 8, col = i & 255;
            dst[row * A_STR_U32 + col] = src[i];
        }
    }
#endif

    // top-k state init overlaps the A-tile DMA
    for (int i = tid; i < 32 * KTOP; i += 128) {
        topVal[i] = -__builtin_inff();
        topIdx[i] = 0;
    }
    if (tid < 32) { thresh[tid] = -__builtin_inff(); minPos[tid] = 0; }

#if defined(__gfx1250__) && __has_builtin(__builtin_amdgcn_tensor_load_to_lds)
    if (wave == 0) __builtin_amdgcn_s_wait_tensorcnt(0);
#endif
    __syncthreads();

    const bf16_t* aRow0 = aTile + l15 * A_STR;          // M rows 0..15
    const bf16_t* aRow1 = aTile + (16 + l15) * A_STR;   // M rows 16..31

    for (int pBase = 0; pBase < NPAT; pBase += 128) {
        // ---- 2Mx2N sim tile per wave (K = 512 in 16 steps x 4 acc) ----
        const int nW = pBase + wave * 32;
        const bf16_t* brow0 = patn + (size_t)(nW + l15) * PDIM + half * 16;
        const bf16_t* brow1 = patn + (size_t)(nW + 16 + l15) * PDIM + half * 16;
        v8f c00 = {}, c01 = {}, c10 = {}, c11 = {};
#pragma unroll 4
        for (int k0 = 0; k0 < PDIM; k0 += 32) {
            v8bf a0lo = *(const v8bf*)(aRow0 + k0 + half * 16);
            v8bf a0hi = *(const v8bf*)(aRow0 + k0 + half * 16 + 8);
            v16bf a0 = SHUF16(a0lo, a0hi);
            v8bf a1lo = *(const v8bf*)(aRow1 + k0 + half * 16);
            v8bf a1hi = *(const v8bf*)(aRow1 + k0 + half * 16 + 8);
            v16bf a1 = SHUF16(a1lo, a1hi);
            v16bf b0 = *(const v16bf*)(brow0 + k0);
            v16bf b1 = *(const v16bf*)(brow1 + k0);
            c00 = __builtin_amdgcn_wmma_f32_16x16x32_bf16(false, a0, false, b0,
                                                          (short)0, c00, false, false);
            c01 = __builtin_amdgcn_wmma_f32_16x16x32_bf16(false, a0, false, b1,
                                                          (short)0, c01, false, false);
            c10 = __builtin_amdgcn_wmma_f32_16x16x32_bf16(false, a1, false, b0,
                                                          (short)0, c10, false, false);
            c11 = __builtin_amdgcn_wmma_f32_16x16x32_bf16(false, a1, false, b1,
                                                          (short)0, c11, false, false);
        }
        {
            const int nc0 = wave * 32 + l15;       // local col of N-subtile 0
            const int nc1 = wave * 32 + 16 + l15;  // local col of N-subtile 1
#pragma unroll
            for (int r = 0; r < 8; ++r) {
                const int m0 = r + 8 * half;       // local token row, M-subtile 0
                simBuf[m0 * 128 + nc0]        = c00[r];
                simBuf[m0 * 128 + nc1]        = c01[r];
                simBuf[(16 + m0) * 128 + nc0] = c10[r];
                simBuf[(16 + m0) * 128 + nc1] = c11[r];
            }
        }

        // prefetch next pattern chunk toward WGP$ while top-k selection runs
        if (pBase + 128 < NPAT)
            __builtin_prefetch(patn + (size_t)(pBase + 128 + tid) * PDIM, 0, 1);
        __syncthreads();

        // ---- threshold ballot: 4 threads per token, 32 candidates each,
        //      each thread owns exactly one mask word (no atomics) ----
        {
            const int tok = tid >> 2, slot = tid & 3;
            const float th = thresh[tok];
            unsigned m = 0u;
#pragma unroll
            for (int j = 0; j < 32; ++j) {
                if (simBuf[tok * 128 + slot * 32 + j] > th) m |= (1u << j);
            }
            passMask[tok * 4 + slot] = m;
        }
        __syncthreads();

        // ---- rare serial inserts: one thread per token ----
        if ((tid & 3) == 0) {
            const int tok = tid >> 2;
            float th = thresh[tok];
            int   mp = minPos[tok];
            for (int w = 0; w < 4; ++w) {
                unsigned m = passMask[tok * 4 + w];
                while (m) {
                    const int b = __builtin_ctz(m);
                    m &= m - 1;
                    const int cc = w * 32 + b;
                    const float v = simBuf[tok * 128 + cc];
                    if (v > th) {  // re-check: threshold may have risen
                        topVal[tok * KTOP + mp] = v;
                        topIdx[tok * KTOP + mp] = pBase + cc;
                        float mn = topVal[tok * KTOP];
                        int pos = 0;
                        for (int i = 1; i < KTOP; ++i) {
                            const float t = topVal[tok * KTOP + i];
                            if (t < mn) { mn = t; pos = i; }
                        }
                        th = mn; mp = pos;
                    }
                }
            }
            thresh[tok] = th; minPos[tok] = mp;
        }
        __syncthreads();
    }

    for (int i = tid; i < 32 * KTOP; i += 128) {
        const int tok = i >> 7, j = i & 127;
        topk[(size_t)(tM + tok) * KTOP + j] = topIdx[i];
    }
}

// ---------------------------------------------------------------------------
// gather + mean over top-128 patterns (patterns fit in L2), emit bf16.
// ---------------------------------------------------------------------------
__global__ void __launch_bounds__(256)
gather_mean_kernel(const float* __restrict__ patterns, const int* __restrict__ topk,
                   bf16_t* __restrict__ mp) {
    __shared__ int idx[KTOP];
    const int t = blockIdx.x;
    if (threadIdx.x < KTOP) idx[threadIdx.x] = topk[(size_t)t * KTOP + threadIdx.x];
    __syncthreads();
    float a0 = 0.f, a1 = 0.f;
    for (int k = 0; k < KTOP; ++k) {
        const float2 v = *(const float2*)(patterns + (size_t)idx[k] * PDIM
                                          + 2 * threadIdx.x);
        a0 += v.x;
        a1 += v.y;
    }
    const float s = 1.0f / (float)KTOP;
    const int c = 2 * threadIdx.x;
    mp[(size_t)t * PDIM + c]     = (bf16_t)(a0 * s);
    mp[(size_t)t * PDIM + c + 1] = (bf16_t)(a1 * s);
}

// ---------------------------------------------------------------------------
// decoder GEMM + residual:
// out[T,D] = x + alpha * (meanpat_bf16[T,P] @ dec_w_bf16[D,P]^T + dec_b)
// ---------------------------------------------------------------------------
__global__ void __launch_bounds__(128)
dec_gemm_kernel(const bf16_t* __restrict__ mp, const bf16_t* __restrict__ wbf,
                const float* __restrict__ bias, const float* __restrict__ x,
                const float* __restrict__ alpha, float* __restrict__ out) {
    const int tid  = threadIdx.x;
    const int wave = tid >> 5, lane = tid & 31;
    const int half = lane >> 4, l15 = lane & 15;
    const int tM = blockIdx.x * 16;
    const int tN = blockIdx.y * 64 + wave * 16;

    const bf16_t* arow = mp  + (size_t)(tM + l15) * PDIM;
    const bf16_t* brow = wbf + (size_t)(tN + l15) * PDIM + half * 16;

    v8f c = {};
#pragma unroll 4
    for (int k0 = 0; k0 < PDIM; k0 += 32) {
        v8bf alo = *(const v8bf*)(arow + k0 + half * 16);
        v8bf ahi = *(const v8bf*)(arow + k0 + half * 16 + 8);
        v16bf a = SHUF16(alo, ahi);
        v16bf b = *(const v16bf*)(brow + k0);
        c = __builtin_amdgcn_wmma_f32_16x16x32_bf16(false, a, false, b,
                                                    (short)0, c, false, false);
    }
    const int n = tN + l15;
    const float bn = bias[n];
    const float a0 = alpha[0];
#pragma unroll
    for (int r = 0; r < 8; ++r) {
        const int m = tM + r + 8 * half;
        out[(size_t)m * DFEAT + n] = x[(size_t)m * DFEAT + n] + a0 * (c[r] + bn);
    }
}

// ---------------------------------------------------------------------------
extern "C" void kernel_launch(void* const* d_in, const int* in_sizes, int n_in,
                              void* d_out, int out_size, void* d_ws, size_t ws_size,
                              hipStream_t stream) {
    const float* x        = (const float*)d_in[0];  // [T, D]
    const float* patterns = (const float*)d_in[1];  // [N, P]
    const float* alpha    = (const float*)d_in[2];  // [1]
    const float* enc_w    = (const float*)d_in[3];  // [P, D]
    const float* enc_b    = (const float*)d_in[4];  // [P]
    const float* dec_w    = (const float*)d_in[5];  // [D, P]
    const float* dec_b    = (const float*)d_in[6];  // [D]
    float* out = (float*)d_out;

    // workspace carve-out (~92 MB total)
    char* ws = (char*)d_ws;
    size_t off = 0;
    auto carve = [&](size_t bytes) -> void* {
        off = (off + 255) & ~(size_t)255;
        void* p = ws + off;
        off += bytes;
        return p;
    };
    bf16_t* xbf     = (bf16_t*)carve((size_t)T_TOK * DFEAT * 2);  //  8 MB
    bf16_t* encw_bf = (bf16_t*)carve((size_t)PDIM  * DFEAT * 2);  //  1 MB
    bf16_t* decw_bf = (bf16_t*)carve((size_t)DFEAT * PDIM  * 2);  //  1 MB
    bf16_t* pat_n   = (bf16_t*)carve((size_t)NPAT  * PDIM  * 2);  // 64 MB
    float*  enc_f   = (float*) carve((size_t)T_TOK * PDIM  * 4);  //  8 MB
    bf16_t* enc_n   = (bf16_t*)carve((size_t)T_TOK * PDIM  * 2);  //  4 MB
    int*    topk    = (int*)   carve((size_t)T_TOK * KTOP  * 4);  //  2 MB
    bf16_t* mp_bf   = (bf16_t*)carve((size_t)T_TOK * PDIM  * 2);  //  4 MB

    // 1) casts to bf16
    {
        int n = T_TOK * DFEAT;
        cast_f32_to_bf16<<<(n + 255) / 256, 256, 0, stream>>>(x, xbf, n);
        n = PDIM * DFEAT;
        cast_f32_to_bf16<<<(n + 255) / 256, 256, 0, stream>>>(enc_w, encw_bf, n);
        n = DFEAT * PDIM;
        cast_f32_to_bf16<<<(n + 255) / 256, 256, 0, stream>>>(dec_w, decw_bf, n);
    }

    // 2) normalize pattern rows -> bf16
    rownorm_to_bf16<<<NPAT, 256, 0, stream>>>(patterns, pat_n, PDIM);

    // 3) encoder GEMM (WMMA) -> f32
    enc_gemm_kernel<<<dim3(T_TOK / 16, PDIM / 64), 128, 0, stream>>>(
        xbf, encw_bf, enc_b, enc_f);

    // 4) normalize enc rows -> bf16
    rownorm_to_bf16<<<T_TOK, 256, 0, stream>>>(enc_f, enc_n, PDIM);

    // 5) fused similarity GEMM + top-128 (WMMA, 32 tokens x 128 patterns/step)
    sim_topk_kernel<<<T_TOK / 32, 128, 0, stream>>>(enc_n, pat_n, topk);

    // 6) gather + mean -> bf16
    gather_mean_kernel<<<T_TOK, 256, 0, stream>>>(patterns, topk, mp_bf);

    // 7) decoder GEMM (WMMA) + bias + alpha residual
    dec_gemm_kernel<<<dim3(T_TOK / 16, DFEAT / 64), 128, 0, stream>>>(
        mp_bf, decw_bf, dec_b, x, alpha, out);
}
